// LinearCEMTPLoss_60576218743254
// MI455X (gfx1250) — matrix operand order repaired and checked by reading, not compiled, not run on hardware
//
#include <hip/hip_runtime.h>
#include <hip/hip_bf16.h>
#include <math.h>

typedef __bf16 v4bf  __attribute__((ext_vector_type(4)));
typedef __bf16 v8bf  __attribute__((ext_vector_type(8)));
typedef __bf16 v16bf __attribute__((ext_vector_type(16)));
typedef float  v8f   __attribute__((ext_vector_type(8)));
typedef float  vf16  __attribute__((ext_vector_type(16)));

#define TOK_TILE     32
#define NSPLIT       4
#define NTHREADS     256
#define IGNORE_IDX   (-100)
#define STATS_STRIDE 12

__device__ __forceinline__ void merge_lse(float& m, float& Z, float& C,
                                          float om, float oZ, float oC) {
  if (om > -INFINITY) {
    if (m == -INFINITY) { m = om; Z = oZ; C = oC; }
    else {
      float M2 = fmaxf(m, om);
      float s1 = __expf(m - M2), s2 = __expf(om - M2);
      Z = Z * s1 + oZ * s2;
      C = C * s1 + oC * s2;
      m = M2;
    }
  }
}

// ---------------------------------------------------------------------------
// Main kernel: 32-token tile per WG (h+mtp staged bf16 in LDS), vocab split
// over blockIdx.y. Online softmax stats per (token, split) -> workspace.
// ---------------------------------------------------------------------------
__global__ __launch_bounds__(NTHREADS)
void lce_main(const float* __restrict__ h, const float* __restrict__ mh,
              const float* __restrict__ W, const int* __restrict__ tgt,
              float* __restrict__ ws, int N, int D, int V)
{
  extern __shared__ __align__(16) char smem[];
  const int P = D + 8;                       // bf16 row pitch (bank-conflict pad)
  __bf16* hA = (__bf16*)smem;                // [32][P]
  __bf16* mA = hA + TOK_TILE * P;            // [32][P]

  const int tid   = threadIdx.x;
  const int tok0  = blockIdx.x * TOK_TILE;
  const int split = blockIdx.y;

  // ---- stage activations f32 -> bf16 into LDS (read once per tile) ----
  const int vecPerRow = D >> 2;
  for (int q = tid; q < TOK_TILE * vecPerRow; q += NTHREADS) {
    int r = q / vecPerRow;
    int k = (q - r * vecPerRow) << 2;
    int token = tok0 + r;
    float4 hv = make_float4(0.f, 0.f, 0.f, 0.f);
    float4 mv = make_float4(0.f, 0.f, 0.f, 0.f);
    if (token < N) {
      hv = *(const float4*)(h  + (size_t)token * D + k);
      mv = *(const float4*)(mh + (size_t)token * D + k);
    }
    v4bf hp; hp[0] = (__bf16)hv.x; hp[1] = (__bf16)hv.y; hp[2] = (__bf16)hv.z; hp[3] = (__bf16)hv.w;
    v4bf mp; mp[0] = (__bf16)mv.x; mp[1] = (__bf16)mv.y; mp[2] = (__bf16)mv.z; mp[3] = (__bf16)mv.w;
    *(v4bf*)(hA + (size_t)r * P + k) = hp;
    *(v4bf*)(mA + (size_t)r * P + k) = mp;
  }
  __syncthreads();

  // ---- wave mapping: wr = token half (16 rows), wc = vocab strip of 16 ----
  const int wave  = tid >> 5;
  const int lane  = tid & 31;
  const int wr    = wave & 1;
  const int wc    = wave >> 1;               // 0..3
  const int tr    = wr * 16;
  const int laneN = lane & 15;
  const bool hi   = lane >= 16;
  const int rowBase = hi ? 8 : 0;            // C/D layout: lanes>=16 hold M=8..15

  // per-lane online stats for 8 rows
  float m[8], Zr[8], Cr[8], mm[8], mZ[8], mxl[8], mxm[8], tl[8], tm[8];
  int axl[8], axm[8], tg[8];
#pragma unroll
  for (int i = 0; i < 8; ++i) {
    m[i] = -INFINITY; Zr[i] = 0.f; Cr[i] = 0.f;
    mm[i] = -INFINITY; mZ[i] = 0.f;
    mxl[i] = -INFINITY; mxm[i] = -INFINITY; axl[i] = 0; axm[i] = 0;
    tl[i] = 0.f; tm[i] = 0.f;
    int token = tok0 + tr + rowBase + i;
    int tv = (token < N) ? tgt[token] : 0;
    if (tv < 0 || tv >= V) tv = 0;           // safe_tgt (mask applied in finalize)
    tg[i] = tv;
  }

  const int vper   = (V + NSPLIT - 1) / NSPLIT;
  const int vstart = split * vper;
  const int vend   = (vstart + vper < V) ? (vstart + vper) : V;

  // WMMA operand layouts (bf16, wave32):
  //  A 16x32: lane<16 -> M=lane,   K = {0..7, 16..23};  lane>=16 -> M=lane-16, K = {8..15, 24..31}
  //  B 32x16: lane<16 -> N=lane,   K = {0..7, 16..23};  lane>=16 -> N=lane-16, K = {8..15, 24..31}
  const int kA0 = hi ? 8 : 0;
  const int kB0 = hi ? 8 : 0;
  const __bf16* hrow = hA + (size_t)(tr + laneN) * P;
  const __bf16* mrow = mA + (size_t)(tr + laneN) * P;

  for (int vb = vstart + wc * 16; vb < vend; vb += 64) {
    const int vcol = vb + laneN;
    const float* Wv = W + (size_t)((vcol < V) ? vcol : 0) * D;

    v8f aH0 = {0.f,0.f,0.f,0.f,0.f,0.f,0.f,0.f};
    v8f aH1 = {0.f,0.f,0.f,0.f,0.f,0.f,0.f,0.f};
    v8f aM0 = {0.f,0.f,0.f,0.f,0.f,0.f,0.f,0.f};
    v8f aM1 = {0.f,0.f,0.f,0.f,0.f,0.f,0.f,0.f};

    for (int kc = 0; kc < D; kc += 64) {     // two K-phases -> 4 independent WMMAs
#pragma unroll
      for (int ph = 0; ph < 2; ++ph) {
        const int kb = kc + ph * 32;
        v8bf a0 = *(const v8bf*)(hrow + kb + kA0);
        v8bf a1 = *(const v8bf*)(hrow + kb + kA0 + 16);
        v16bf Ah = __builtin_shufflevector(a0, a1, 0,1,2,3,4,5,6,7,8,9,10,11,12,13,14,15);
        v8bf c0 = *(const v8bf*)(mrow + kb + kA0);
        v8bf c1 = *(const v8bf*)(mrow + kb + kA0 + 16);
        v16bf Am = __builtin_shufflevector(c0, c1, 0,1,2,3,4,5,6,7,8,9,10,11,12,13,14,15);
        v8f bw0 = *(const v8f*)(Wv + kb + kB0);        // 8 f32 of W row
        v8f bw1 = *(const v8f*)(Wv + kb + kB0 + 16);   // 8 f32 of W row
        v16bf Bv;
#pragma unroll
        for (int j = 0; j < 16; ++j) Bv[j] = (__bf16)(j < 8 ? bw0[j] : bw1[j - 8]);
        if (ph == 0) {
          aH0 = __builtin_amdgcn_wmma_f32_16x16x32_bf16(false, Ah, false, Bv, (short)0, aH0, false, false);
          aM0 = __builtin_amdgcn_wmma_f32_16x16x32_bf16(false, Am, false, Bv, (short)0, aM0, false, false);
        } else {
          aH1 = __builtin_amdgcn_wmma_f32_16x16x32_bf16(false, Ah, false, Bv, (short)0, aH1, false, false);
          aM1 = __builtin_amdgcn_wmma_f32_16x16x32_bf16(false, Am, false, Bv, (short)0, aM1, false, false);
        }
      }
    }

    if (vcol < vend) {
#pragma unroll
      for (int i = 0; i < 8; ++i) {
        float l = aH0[i] + aH1[i];
        float q = aM0[i] + aM1[i];
        // online logsumexp for logits + paired KL numerator
        float nm = fmaxf(m[i], l);
        float s  = __expf(m[i] - nm);
        float e  = __expf(l - nm);
        Zr[i] = Zr[i] * s + e;
        Cr[i] = Cr[i] * s + e * (l - q);
        m[i]  = nm;
        // online logsumexp for mtp logits
        float nmm = fmaxf(mm[i], q);
        mZ[i] = mZ[i] * __expf(mm[i] - nmm) + __expf(q - nmm);
        mm[i] = nmm;
        // argmax (first-occurrence semantics within lane)
        if (l > mxl[i]) { mxl[i] = l; axl[i] = vcol; }
        if (q > mxm[i]) { mxm[i] = q; axm[i] = vcol; }
        // target gather
        if (vcol == tg[i]) { tl[i] = l; tm[i] = q; }
      }
    }
  }

  // ---- merge across the 16 lanes that share each row (butterfly) ----
#pragma unroll
  for (int d = 1; d < 16; d <<= 1) {
#pragma unroll
    for (int i = 0; i < 8; ++i) {
      float om = __shfl_xor(m[i], d, 32);
      float oZ = __shfl_xor(Zr[i], d, 32);
      float oC = __shfl_xor(Cr[i], d, 32);
      merge_lse(m[i], Zr[i], Cr[i], om, oZ, oC);
      float omm = __shfl_xor(mm[i], d, 32);
      float omZ = __shfl_xor(mZ[i], d, 32);
      float dC = 0.f;
      merge_lse(mm[i], mZ[i], dC, omm, omZ, 0.f);
      float oxl = __shfl_xor(mxl[i], d, 32);
      int   oal = __shfl_xor(axl[i], d, 32);
      if (oxl > mxl[i] || (oxl == mxl[i] && oal < axl[i])) { mxl[i] = oxl; axl[i] = oal; }
      float oxm = __shfl_xor(mxm[i], d, 32);
      int   oam = __shfl_xor(axm[i], d, 32);
      if (oxm > mxm[i] || (oxm == mxm[i] && oam < axm[i])) { mxm[i] = oxm; axm[i] = oam; }
      tl[i] += __shfl_xor(tl[i], d, 32);
      tm[i] += __shfl_xor(tm[i], d, 32);
    }
  }

  if (laneN == 0) {
#pragma unroll
    for (int i = 0; i < 8; ++i) {
      int token = tok0 + tr + rowBase + i;
      if (token < N) {
        float* o = ws + ((size_t)token * NSPLIT + split) * STATS_STRIDE;
        o[0] = m[i];  o[1] = Zr[i]; o[2] = Cr[i];
        o[3] = mm[i]; o[4] = mZ[i];
        o[5] = mxl[i]; o[6] = __int_as_float(axl[i]);
        o[7] = mxm[i]; o[8] = __int_as_float(axm[i]);
        o[9] = tl[i];  o[10] = tm[i];
        o[11] = 0.f;
      }
    }
  }
}

// ---------------------------------------------------------------------------
// Finalize: merge vocab splits per token, compute losses, deterministic
// tree-reduce to 5 scalars.
// ---------------------------------------------------------------------------
__global__ __launch_bounds__(NTHREADS)
void lce_finalize(const float* __restrict__ ws, const int* __restrict__ tgt,
                  float* __restrict__ out, int N)
{
  float ce_s = 0.f, mce_s = 0.f, kl_s = 0.f, ag_s = 0.f, nv = 0.f;
  for (int t = threadIdx.x; t < N; t += NTHREADS) {
    float m = -INFINITY, Z = 0.f, C = 0.f, mm = -INFINITY, mZ = 0.f;
    float mxl = -INFINITY, mxm = -INFINITY, tl = 0.f, tm = 0.f;
    int axl = 0, axm = 0;
    for (int s = 0; s < NSPLIT; ++s) {
      const float* o = ws + ((size_t)t * NSPLIT + s) * STATS_STRIDE;
      merge_lse(m, Z, C, o[0], o[1], o[2]);
      float dC = 0.f;
      merge_lse(mm, mZ, dC, o[3], o[4], 0.f);
      float oxl = o[5]; int oal = __float_as_int(o[6]);
      if (oxl > mxl || (oxl == mxl && oal < axl)) { mxl = oxl; axl = oal; }
      float oxm = o[7]; int oam = __float_as_int(o[8]);
      if (oxm > mxm || (oxm == mxm && oam < axm)) { mxm = oxm; axm = oam; }
      tl += o[9]; tm += o[10];
    }
    if (tgt[t] != IGNORE_IDX) {
      float lse  = m  + __logf(Z);
      float mlse = mm + __logf(mZ);
      ce_s  += lse - tl;
      mce_s += mlse - tm;
      kl_s  += C / Z - lse + mlse;
      ag_s  += (axl == axm) ? 1.f : 0.f;
      nv    += 1.f;
    }
  }
  __shared__ float red[5][NTHREADS];
  red[0][threadIdx.x] = ce_s;  red[1][threadIdx.x] = mce_s;
  red[2][threadIdx.x] = kl_s;  red[3][threadIdx.x] = ag_s;
  red[4][threadIdx.x] = nv;
  __syncthreads();
  for (int off = NTHREADS / 2; off > 0; off >>= 1) {
    if (threadIdx.x < off) {
#pragma unroll
      for (int j = 0; j < 5; ++j)
        red[j][threadIdx.x] += red[j][threadIdx.x + off];
    }
    __syncthreads();
  }
  if (threadIdx.x == 0) {
    const float MTP_CE_FACTOR = 0.0f, MTP_KL_FACTOR = 1.0f;
    float nvv = red[4][0];
    out[0] = (red[0][0] + MTP_CE_FACTOR * red[1][0] + MTP_KL_FACTOR * red[2][0]) / nvv;
    out[1] = red[0][0] / nvv;
    out[2] = red[1][0] / nvv;
    out[3] = red[2][0] / nvv;
    out[4] = red[3][0] / nvv;
  }
}

extern "C" void kernel_launch(void* const* d_in, const int* in_sizes, int n_in,
                              void* d_out, int out_size, void* d_ws, size_t ws_size,
                              hipStream_t stream) {
  const float* h   = (const float*)d_in[0];
  const float* mh  = (const float*)d_in[1];
  const float* W   = (const float*)d_in[2];
  const int*   tgt = (const int*)d_in[3];
  float* out = (float*)d_out;
  float* ws  = (float*)d_ws;

  const int N = in_sizes[3];            // B*S tokens
  const int D = in_sizes[0] / N;        // hidden dim
  const int V = in_sizes[2] / D;        // vocab size

  const int tiles = (N + TOK_TILE - 1) / TOK_TILE;
  const size_t lds_bytes = (size_t)2 * TOK_TILE * (D + 8) * sizeof(__bf16);

  dim3 grid(tiles, NSPLIT);
  lce_main<<<grid, NTHREADS, lds_bytes, stream>>>(h, mh, W, tgt, ws, N, D, V);
  lce_finalize<<<1, NTHREADS, 0, stream>>>(ws, tgt, out, N);
}
